// GraphNet_47072841564373
// MI455X (gfx1250) — matrix operand order repaired and checked by reading, not compiled, not run on hardware
//
#include <hip/hip_runtime.h>

#define B_   1024
#define N_   30
#define P_   16
#define NR_  870
#define DR_  3
#define DE_  5
#define DO_  6
#define NT_  5
#define HID_ 128

typedef __attribute__((ext_vector_type(16))) _Float16 v16h;
typedef __attribute__((ext_vector_type(8)))  float    v8f;

// ---------------------------------------------------------------- shared mem
// ~149 KB -> two workgroups co-resident per 320 KB WGP (4 waves / SIMD32).
struct __align__(16) Smem {
    _Float16 wh1[128 * 64];     // fr1_w [128][35 -> 64]   (reused 55x/block)
    _Float16 wh2[64 * 128];     // fr2_w [64][128]
    _Float16 wh3[16 * 64];      // fr3_w [5->16][64]
    _Float16 xh [16 * 32];      // x^T: [p][n pad 32]  (column-major B for Bm)
    _Float16 act0[8 * 16 * 64]; // per-wave edge-MLP input  [16][64]
    _Float16 act1[8 * 16 * 128];// per-wave edge hidden1    [16][128]
    _Float16 act2[8 * 16 * 64]; // per-wave edge hidden2    [16][64]
    _Float16 Et [16 * 896];     // E^T  [e pad16][r pad 896]
    _Float16 actO [32 * 32];    // object input [n pad32][21 -> 32]
    _Float16 actO1[32 * 128];
    _Float16 actO2[32 * 64];
    float ebar[16 * 32];        // [e][n] f32 accumulators
    float fvec[192];            // O flattened [180 -> 192]
    float h1[128];
    float h2[64];
    float b1[128], b2[64], b3[16];   // edge biases (hot, 55x reuse)
};

// ---------------------------------------------------------------- wmma helper
__device__ __forceinline__ v8f wmma_f16(v16h a, v16h b, v8f c) {
    return __builtin_amdgcn_wmma_f32_16x16x32_f16(
        /*neg_a=*/false, a, /*neg_b=*/false, b,
        /*c_mod=*/(short)0, c, /*reuse_a=*/false, /*reuse_b=*/false);
}

// A-fragment (16x32, f16) from LDS row-major activations; base points at (row0,k0).
__device__ __forceinline__ v16h load_a_lds(const _Float16* base, int ld, int hf, int l15) {
    const _Float16* row = base + l15 * ld;
    v16h a;
#pragma unroll
    for (int i = 0; i < 8; ++i) {
        a[i]     = row[hf * 16 + i];       // K block {0..7} / {16..23}
        a[8 + i] = row[hf * 16 + 8 + i];   // K block {8..15} / {24..31}
    }
    return a;
}

// B-fragment (32x16, f16) from LDS; W stored [out][K] row-major == column-major B.
// wbase points at W[o0][k0].
__device__ __forceinline__ v16h load_b_lds(const _Float16* wbase, int ldk, int hf, int l15) {
    const _Float16* p = wbase + l15 * ldk + hf * 16;
    v16h b;
#pragma unroll
    for (int i = 0; i < 16; ++i) b[i] = p[i];
    return b;
}

// B-fragment straight from a global f32 weight matrix W[O][I] (row-major),
// with zero padding outside [O, I].  Rows are contiguous -> b128-friendly.
__device__ __forceinline__ v16h load_b_gw(const float* __restrict__ W,
                                          int O, int I, int o0, int k0,
                                          int hf, int l15) {
    const int o = o0 + l15;
    const int kb = k0 + hf * 16;
    v16h b;
    if (o < O && kb + 16 <= I) {
        const float* p = W + o * I + kb;
#pragma unroll
        for (int i = 0; i < 16; ++i) b[i] = (_Float16)p[i];
    } else {
#pragma unroll
        for (int i = 0; i < 16; ++i) {
            const int k = kb + i;
            b[i] = (o < O && k < I) ? (_Float16)W[o * I + k] : (_Float16)0.f;
        }
    }
    return b;
}

// A-fragment for Bm: A[r][n] = M[n][r0+r], M = RR or RS slice of one batch.
__device__ __forceinline__ v16h load_a_edge(const float* Mb, int r0, int hf, int l15) {
    const int r = r0 + l15;
    const bool rok = (r < NR_);
    v16h a;
#pragma unroll
    for (int i = 0; i < 8; ++i) {
        const int n0 = hf * 16 + i;         // < 24  (< N_ always)
        a[i] = rok ? (_Float16)Mb[n0 * NR_ + r] : (_Float16)0.f;
        const int n1 = hf * 16 + 8 + i;     // 8..15 / 24..31
        a[8 + i] = (rok && n1 < N_) ? (_Float16)Mb[n1 * NR_ + r] : (_Float16)0.f;
    }
    return a;
}

// B-fragment for Ebar: B[r][n] = RR[n][r]; rows of RR are contiguous in global.
// Fast unguarded path for fully in-bounds K chunks (all but the tail).
__device__ __forceinline__ v16h load_b_rrT(const float* Mb, int k0, int ntile, int hf, int l15) {
    const int node = ntile * 16 + l15;
    const int rb = k0 + hf * 16;
    v16h b;
    if (node < N_ && rb + 16 <= NR_) {
        const float* p = Mb + node * NR_ + rb;
#pragma unroll
        for (int i = 0; i < 16; ++i) b[i] = (_Float16)p[i];
    } else {
#pragma unroll
        for (int i = 0; i < 16; ++i) {
            const int r = rb + i;
            b[i] = (node < N_ && r < NR_) ? (_Float16)Mb[node * NR_ + r] : (_Float16)0.f;
        }
    }
    return b;
}

// Fused bias+ReLU store of a C/D fragment to LDS f16 activations (bias in LDS).
__device__ __forceinline__ void store_c_relu(_Float16* dst, int ld, v8f d,
                                             const float* bias, int o0, int hf, int l15) {
    const float bv = bias[o0 + l15];
#pragma unroll
    for (int v = 0; v < 8; ++v) {
        float x = d[v] + bv;
        x = x > 0.f ? x : 0.f;
        dst[(v + 8 * hf) * ld + l15] = (_Float16)x;
    }
}

// ---------------------------------------------------------------- staging
__device__ __forceinline__ void stage_w(_Float16* dst, const float* src,
                                        int O, int Opad, int I, int Ipad, int tid) {
    const int tot = Opad * Ipad;
    for (int idx = tid; idx < tot; idx += 256) {
        const int o = idx / Ipad;
        const int k = idx - o * Ipad;
        dst[idx] = (o < O && k < I) ? (_Float16)src[o * I + k] : (_Float16)0.f;
    }
}
__device__ __forceinline__ void stage_bias(float* dst, const float* src,
                                           int Nv, int Npad, int tid) {
    for (int i = tid; i < Npad; i += 256) dst[i] = (i < Nv) ? src[i] : 0.f;
}
__device__ __forceinline__ void zero_h(_Float16* dst, int n, int tid) {
    for (int i = tid; i < n; i += 256) dst[i] = (_Float16)0.f;
}
__device__ __forceinline__ void zero_f(float* dst, int n, int tid) {
    for (int i = tid; i < n; i += 256) dst[i] = 0.f;
}

// ---------------------------------------------------------------- kernel
__global__ void __launch_bounds__(256, 1)
inet_kernel(const float* __restrict__ x,   const float* __restrict__ RR,
            const float* __restrict__ RS,  const float* __restrict__ RA,
            const float* __restrict__ fr1_w, const float* __restrict__ fr1_b,
            const float* __restrict__ fr2_w, const float* __restrict__ fr2_b,
            const float* __restrict__ fr3_w, const float* __restrict__ fr3_b,
            const float* __restrict__ fo1_w, const float* __restrict__ fo1_b,
            const float* __restrict__ fo2_w, const float* __restrict__ fo2_b,
            const float* __restrict__ fo3_w, const float* __restrict__ fo3_b,
            const float* __restrict__ fc1_w, const float* __restrict__ fc1_b,
            const float* __restrict__ fc2_w, const float* __restrict__ fc2_b,
            const float* __restrict__ fc3_w, const float* __restrict__ fc3_b,
            float* __restrict__ out)
{
    extern __shared__ char smem_raw[];
    Smem* sm = reinterpret_cast<Smem*>(smem_raw);

    const int tid  = threadIdx.x;
    const int wave = tid >> 5;
    const int lane = tid & 31;
    const int hf   = lane >> 4;
    const int l15  = lane & 15;
    const int b    = blockIdx.x;

    const float* xb  = x  + b * (N_ * P_);
    const float* RRb = RR + b * (N_ * NR_);
    const float* RSb = RS + b * (N_ * NR_);
    const float* RAb = RA + b * (DR_ * NR_);

    // ---------------- phase 0: stage hot (edge) weights, x, zero scratch ----
    stage_w(sm->wh1, fr1_w, 128, 128, 2 * P_ + DR_, 64,  tid);
    stage_w(sm->wh2, fr2_w, 64,  64,  128,          128, tid);
    stage_w(sm->wh3, fr3_w, DE_, 16,  64,           64,  tid);
    stage_bias(sm->b1,  fr1_b, 128, 128, tid);
    stage_bias(sm->b2,  fr2_b, 64,  64,  tid);
    stage_bias(sm->b3,  fr3_b, DE_, 16,  tid);
    for (int idx = tid; idx < 16 * 32; idx += 256) {   // xh[p][n pad 32]
        const int p = idx >> 5, n = idx & 31;
        sm->xh[idx] = (n < N_) ? (_Float16)xb[n * P_ + p] : (_Float16)0.f;
    }
    zero_h(sm->Et,   16 * 896,    tid);
    zero_h(sm->act0, 8 * 16 * 64, tid);   // keeps cols 35..63 zero forever
    zero_f(sm->ebar, 16 * 32,     tid);
    zero_f(sm->fvec, 192,         tid);
    __syncthreads();

    // ---------------- phase 1: edge pipeline, 16-edge tiles per wave --------
    _Float16* a0 = sm->act0 + wave * (16 * 64);
    _Float16* a1 = sm->act1 + wave * (16 * 128);
    _Float16* a2 = sm->act2 + wave * (16 * 64);

    for (int t = wave; t < 55; t += 8) {
        const int r0 = t * 16;
        // Bm tile: Orr = RRtile^T @ x, Ors = RStile^T @ x   (K = 30 pad 32)
        v16h aRR = load_a_edge(RRb, r0, hf, l15);
        v16h aRS = load_a_edge(RSb, r0, hf, l15);
        v16h bx  = load_b_lds(sm->xh, 32, hf, l15);
        v8f z = {};
        v8f dOrr = wmma_f16(aRR, bx, z);
        v8f dOrs = wmma_f16(aRS, bx, z);
#pragma unroll
        for (int v = 0; v < 8; ++v) {
            const int m = v + 8 * hf;
            a0[m * 64 + l15]      = (_Float16)dOrr[v];
            a0[m * 64 + 16 + l15] = (_Float16)dOrs[v];
        }
        if (hf == 0) {                       // RA features, row m = l15
            const int rr = r0 + l15;
#pragma unroll
            for (int d = 0; d < DR_; ++d)
                a0[l15 * 64 + 32 + d] =
                    (rr < NR_) ? (_Float16)RAb[d * NR_ + rr] : (_Float16)0.f;
        }

        // edge layer 1: [16,64(35)] @ [64,128]
#pragma unroll
        for (int j = 0; j < 8; ++j) {
            v8f acc = {};
#pragma unroll
            for (int kc = 0; kc < 2; ++kc) {
                v16h av = load_a_lds(a0 + kc * 32, 64, hf, l15);
                v16h bv = load_b_lds(sm->wh1 + (j * 16) * 64 + kc * 32, 64, hf, l15);
                acc = wmma_f16(av, bv, acc);
            }
            store_c_relu(a1, 128, acc, sm->b1, j * 16, hf, l15);
        }
        // edge layer 2: [16,128] @ [128,64]
#pragma unroll
        for (int j = 0; j < 4; ++j) {
            v8f acc = {};
#pragma unroll
            for (int kc = 0; kc < 4; ++kc) {
                v16h av = load_a_lds(a1 + kc * 32, 128, hf, l15);
                v16h bv = load_b_lds(sm->wh2 + (j * 16) * 128 + kc * 32, 128, hf, l15);
                acc = wmma_f16(av, bv, acc);
            }
            store_c_relu(a2, 64, acc, sm->b2, j * 16, hf, l15);
        }
        // edge layer 3: [16,64] @ [64,16(5)] -> E, scatter to Et[e][r]
        {
            v8f acc = {};
#pragma unroll
            for (int kc = 0; kc < 2; ++kc) {
                v16h av = load_a_lds(a2 + kc * 32, 64, hf, l15);
                v16h bv = load_b_lds(sm->wh3 + kc * 32, 64, hf, l15);
                acc = wmma_f16(av, bv, acc);
            }
            const float bv3 = sm->b3[l15];
#pragma unroll
            for (int v = 0; v < 8; ++v) {
                const int rr = r0 + v + 8 * hf;
                float e = acc[v] + bv3;
                e = e > 0.f ? e : 0.f;
                sm->Et[l15 * 896 + rr] = (rr < NR_) ? (_Float16)e : (_Float16)0.f;
            }
        }
    }
    __syncthreads();

    // ---------------- phase 2: Ebar = E^T @ RR^T  (M=16(5), N=32(30), K=896) -
    {
        const int ntile = wave >> 2;      // 0..1 (node halves)
        const int kg    = wave & 3;       // 0..3 (7 K-chunks each, 28 total)
        v8f acc = {};
        for (int kc = kg * 7; kc < kg * 7 + 7; ++kc) {
            const int k0 = kc * 32;
            v16h av = load_a_lds(sm->Et + k0, 896, hf, l15);
            v16h bv = load_b_rrT(RRb, k0, ntile, hf, l15);
            acc = wmma_f16(av, bv, acc);
        }
        const int node = ntile * 16 + l15;
#pragma unroll
        for (int v = 0; v < 8; ++v)
            atomicAdd(&sm->ebar[(v + 8 * hf) * 32 + node], acc[v]);
    }
    __syncthreads();

    // ---------------- phase 3: object MLP input [n pad32][21 -> 32] ---------
    for (int idx = tid; idx < 32 * 32; idx += 256) {
        const int n = idx >> 5, k = idx & 31;
        float v = 0.f;
        if (n < N_) {
            if (k < P_)            v = xb[n * P_ + k];
            else if (k < P_ + DE_) v = sm->ebar[(k - P_) * 32 + n];
        }
        sm->actO[idx] = (_Float16)v;
    }
    __syncthreads();

    // object layer 1: [32,32(21)] @ [32,128]; weights straight from global f32
    for (int c = wave; c < 16; c += 8) {
        const int mt = c & 1, j = c >> 1;
        v16h av = load_a_lds(sm->actO + mt * 16 * 32, 32, hf, l15);
        v16h bv = load_b_gw(fo1_w, HID_, P_ + DE_, j * 16, 0, hf, l15);
        v8f z = {};
        v8f acc = wmma_f16(av, bv, z);
        const float bv1 = fo1_b[j * 16 + l15];
#pragma unroll
        for (int v = 0; v < 8; ++v) {
            float xv = acc[v] + bv1;
            xv = xv > 0.f ? xv : 0.f;
            sm->actO1[(mt * 16 + v + 8 * hf) * 128 + j * 16 + l15] = (_Float16)xv;
        }
    }
    __syncthreads();

    // object layer 2: [32,128] @ [128,64]
    {
        const int mt = wave & 1, j = wave >> 1;   // 8 combos, one per wave
        v8f acc = {};
#pragma unroll
        for (int kc = 0; kc < 4; ++kc) {
            v16h av = load_a_lds(sm->actO1 + mt * 16 * 128 + kc * 32, 128, hf, l15);
            v16h bv = load_b_gw(fo2_w, 64, HID_, j * 16, kc * 32, hf, l15);
            acc = wmma_f16(av, bv, acc);
        }
        const float bv2 = fo2_b[j * 16 + l15];
#pragma unroll
        for (int v = 0; v < 8; ++v) {
            float xv = acc[v] + bv2;
            xv = xv > 0.f ? xv : 0.f;
            sm->actO2[(mt * 16 + v + 8 * hf) * 64 + j * 16 + l15] = (_Float16)xv;
        }
    }
    __syncthreads();

    // object layer 3: [32,64] @ [64,16(6)] -> O, flatten to fvec[n*6+o]
    if (wave < 2) {
        const int mt = wave;
        v8f acc = {};
#pragma unroll
        for (int kc = 0; kc < 2; ++kc) {
            v16h av = load_a_lds(sm->actO2 + mt * 16 * 64 + kc * 32, 64, hf, l15);
            v16h bv = load_b_gw(fo3_w, DO_, 64, 0, kc * 32, hf, l15);
            acc = wmma_f16(av, bv, acc);
        }
        const float bv3 = (l15 < DO_) ? fo3_b[l15] : 0.f;
#pragma unroll
        for (int v = 0; v < 8; ++v) {
            const int n = mt * 16 + v + 8 * hf;
            float o = acc[v] + bv3;
            o = o > 0.f ? o : 0.f;
            if (n < N_ && l15 < DO_) sm->fvec[n * DO_ + l15] = o;
        }
    }
    __syncthreads();

    // ---------------- phase 4: fc head (tiny, f32 VALU) ---------------------
    if (tid < 128) {
        float s = fc1_b[tid];
        const float* wr = fc1_w + tid * (DO_ * N_);
        for (int k = 0; k < DO_ * N_; ++k) s += sm->fvec[k] * wr[k];
        sm->h1[tid] = s > 0.f ? s : 0.f;
    }
    __syncthreads();
    if (tid < 64) {
        float s = fc2_b[tid];
        const float* wr = fc2_w + tid * 128;
        for (int k = 0; k < 128; ++k) s += sm->h1[k] * wr[k];
        sm->h2[tid] = s > 0.f ? s : 0.f;
    }
    __syncthreads();
    if (tid < NT_) {
        float s = fc3_b[tid];
        const float* wr = fc3_w + tid * 64;
        for (int k = 0; k < 64; ++k) s += sm->h2[k] * wr[k];
        out[b * NT_ + tid] = s;
    }
}

// ---------------------------------------------------------------- launcher
extern "C" void kernel_launch(void* const* d_in, const int* in_sizes, int n_in,
                              void* d_out, int out_size, void* d_ws, size_t ws_size,
                              hipStream_t stream) {
    (void)in_sizes; (void)n_in; (void)out_size; (void)d_ws; (void)ws_size;
    const float* x     = (const float*)d_in[0];
    const float* RR    = (const float*)d_in[1];
    const float* RS    = (const float*)d_in[2];
    const float* RA    = (const float*)d_in[3];
    const float* fr1_w = (const float*)d_in[4];
    const float* fr1_b = (const float*)d_in[5];
    const float* fr2_w = (const float*)d_in[6];
    const float* fr2_b = (const float*)d_in[7];
    const float* fr3_w = (const float*)d_in[8];
    const float* fr3_b = (const float*)d_in[9];
    const float* fo1_w = (const float*)d_in[10];
    const float* fo1_b = (const float*)d_in[11];
    const float* fo2_w = (const float*)d_in[12];
    const float* fo2_b = (const float*)d_in[13];
    const float* fo3_w = (const float*)d_in[14];
    const float* fo3_b = (const float*)d_in[15];
    const float* fc1_w = (const float*)d_in[16];
    const float* fc1_b = (const float*)d_in[17];
    const float* fc2_w = (const float*)d_in[18];
    const float* fc2_b = (const float*)d_in[19];
    const float* fc3_w = (const float*)d_in[20];
    const float* fc3_b = (const float*)d_in[21];
    float* out = (float*)d_out;

    const size_t smem_bytes = sizeof(Smem);
    hipFuncSetAttribute((const void*)inet_kernel,
                        hipFuncAttributeMaxDynamicSharedMemorySize,
                        (int)smem_bytes);
    inet_kernel<<<dim3(B_), dim3(256), smem_bytes, stream>>>(
        x, RR, RS, RA,
        fr1_w, fr1_b, fr2_w, fr2_b, fr3_w, fr3_b,
        fo1_w, fo1_b, fo2_w, fo2_b, fo3_w, fo3_b,
        fc1_w, fc1_b, fc2_w, fc2_b, fc3_w, fc3_b,
        out);
}
